// GraphNN_39651138077401
// MI455X (gfx1250) — matrix operand order repaired and checked by reading, not compiled, not run on hardware
//
#include <hip/hip_runtime.h>
#include <hip/hip_bf16.h>
#include <math.h>

// ---------------------------------------------------------------------------
// Problem constants (from reference): B=8, N=1000, E=4000, H=256, 2 hops.
// Padded dims so every GEMM is an exact multiple of the 128x128x32 block tile.
// ---------------------------------------------------------------------------
#define BATCH   8
#define N_REAL  1000
#define E_REAL  4000
#define H_DIM   256
#define N_P     1024
#define E_P     4096

typedef __attribute__((ext_vector_type(16))) __bf16 bf16x16;
typedef __attribute__((ext_vector_type(8)))  __bf16 bf16x8;
typedef __attribute__((ext_vector_type(8)))  float  f32x8;

__device__ __forceinline__ float sigmoidf_(float x) { return 1.0f / (1.0f + expf(-x)); }

// LDS byte address of a shared-memory pointer: generic addresses in the LDS
// aperture carry the LDS offset in addr[31:0] (ISA 10.2), so truncation works.
__device__ __forceinline__ unsigned lds_addr32(const void* p) {
  return (unsigned)(unsigned long long)p;
}

// Async global -> LDS 128-bit copy (tracked by ASYNCcnt, no VGPR round-trip).
__device__ __forceinline__ void async_copy_b128(unsigned lds_addr, const void* gaddr) {
  asm volatile("global_load_async_to_lds_b128 %0, %1, off"
               :: "v"(lds_addr), "v"(gaddr)
               : "memory");
}

__device__ __forceinline__ void wait_asynccnt0() {
  asm volatile("s_wait_asynccnt 0x0" ::: "memory");
}

// ---------------------------------------------------------------------------
// Pack / convert kernels: f32 -> bf16 (or f32) with zero padding.
// grid.z = batch; src is [rows, cols] per batch, dst is [rows_p, cols_p].
// ---------------------------------------------------------------------------
__global__ void pack_bf16(const float* __restrict__ src, __bf16* __restrict__ dst,
                          int rows, int cols, int rows_p, int cols_p) {
  const int b = blockIdx.z;
  const size_t total = (size_t)rows_p * cols_p;
  const size_t idx = (size_t)blockIdx.x * blockDim.x + threadIdx.x;
  if (idx >= total) return;
  const int r = (int)(idx / cols_p);
  const int c = (int)(idx % cols_p);
  float v = 0.0f;
  if (r < rows && c < cols)
    v = src[(size_t)b * rows * cols + (size_t)r * cols + c];
  dst[(size_t)b * total + idx] = (__bf16)v;
}

__global__ void pack_f32(const float* __restrict__ src, float* __restrict__ dst,
                         int rows, int cols, int rows_p, int cols_p) {
  const int b = blockIdx.z;
  const size_t total = (size_t)rows_p * cols_p;
  const size_t idx = (size_t)blockIdx.x * blockDim.x + threadIdx.x;
  if (idx >= total) return;
  const int r = (int)(idx / cols_p);
  const int c = (int)(idx % cols_p);
  float v = 0.0f;
  if (r < rows && c < cols)
    v = src[(size_t)b * rows * cols + (size_t)r * cols + c];
  dst[(size_t)b * total + idx] = v;
}

// ---------------------------------------------------------------------------
// bf16 WMMA GEMM:  C[M,N] = A * B, f32 accumulate.
//   TA=false: A is [M,K] row-major.  TA=true: A is [K,M] row-major (computes A^T * B).
//   TB=false: B is [K,N] row-major.  TB=true: B is [N,K] row-major (computes A * B^T).
// Block: 256 threads = 8 wave32 waves; tile 128x128x32; wave tile 64x32 (4x2 WMMA).
// Contiguous staging paths use GLOBAL_LOAD_ASYNC_TO_LDS_B128 (ASYNCcnt);
// transposing paths use two-phase register staging (loads pipelined).
// M, N, K must be multiples of 128/128/32 (guaranteed by padding).
// ---------------------------------------------------------------------------
#define BM 128
#define BN 128
#define BK 32
#define LDS_K 40   // 32 + 8 pad (keeps 16B alignment: 40*2 = 80 = 5*16)

__device__ __forceinline__ bf16x16 ld_afrag(const __bf16* p) {
  // A frag: lanes<16 hold K{kb..kb+7, kb+16..kb+23}, kb = 0 (lanes<16) or 8.
  bf16x8 lo = *(const bf16x8*)(p);
  bf16x8 hi = *(const bf16x8*)(p + 16);
  bf16x16 f;
#pragma unroll
  for (int i = 0; i < 8; ++i) { f[i] = lo[i]; f[i + 8] = hi[i]; }
  return f;
}

__device__ __forceinline__ bf16x16 ld_bfrag(const __bf16* p) {
  // B frag: column per lane; lanes<16 hold K{0..7,16..23}, lanes>=16 hold K{8..15,24..31}.
  bf16x8 lo = *(const bf16x8*)(p);
  bf16x8 hi = *(const bf16x8*)(p + 16);
  bf16x16 f;
#pragma unroll
  for (int i = 0; i < 8; ++i) { f[i] = lo[i]; f[i + 8] = hi[i]; }
  return f;
}

template <bool TA, bool TB, typename OUT_T>
__global__ __launch_bounds__(256) void gemm_wmma(
    const __bf16* __restrict__ A, const __bf16* __restrict__ B, OUT_T* __restrict__ C,
    int M, int N, int K, int lda, int ldb, int ldc,
    size_t strideA, size_t strideB, size_t strideC) {
  __shared__ __align__(16) __bf16 As[BM * LDS_K];
  __shared__ __align__(16) __bf16 Bs[BN * LDS_K];

  const int tid  = threadIdx.x;
  const int wave = tid >> 5;
  const int lane = tid & 31;
  const int wm   = wave >> 2;          // 0..1  (64-row sub-tile)
  const int wn   = wave & 3;           // 0..3  (32-col sub-tile)
  const int lr   = lane & 15;
  const int kb   = (lane >> 4) << 3;   // 0 or 8  (A-frag K base)
  const int khB  = (lane >> 4) << 4;   // 0 or 16 (unused)
  const int m0   = blockIdx.y * BM;
  const int n0   = blockIdx.x * BN;

  A += (size_t)blockIdx.z * strideA;
  B += (size_t)blockIdx.z * strideB;
  C += (size_t)blockIdx.z * strideC;

  f32x8 acc[4][2];
#pragma unroll
  for (int mi = 0; mi < 4; ++mi)
#pragma unroll
    for (int ni = 0; ni < 2; ++ni)
      acc[mi][ni] = (f32x8){0.f, 0.f, 0.f, 0.f, 0.f, 0.f, 0.f, 0.f};

  for (int k0 = 0; k0 < K; k0 += BK) {
    // --- stage A tile into LDS as As[m][k] (2 x 128bit per thread) ---
    if (!TA) {
#pragma unroll
      for (int i = 0; i < 2; ++i) {
        const int c = tid + i * 256;
        const int m = c >> 2, k8 = (c & 3) << 3;
        async_copy_b128(lds_addr32(&As[m * LDS_K + k8]),
                        A + (size_t)(m0 + m) * lda + k0 + k8);
      }
    } else {
      bf16x8 va[2];
#pragma unroll
      for (int i = 0; i < 2; ++i) {
        const int c = tid + i * 256;
        const int k = c >> 4, m8 = (c & 15) << 3;
        va[i] = *(const bf16x8*)(A + (size_t)(k0 + k) * lda + m0 + m8);
      }
#pragma unroll
      for (int i = 0; i < 2; ++i) {
        const int c = tid + i * 256;
        const int k = c >> 4, m8 = (c & 15) << 3;
#pragma unroll
        for (int j = 0; j < 8; ++j) As[(m8 + j) * LDS_K + k] = va[i][j];
      }
    }
    // --- stage B tile into LDS as Bs[n][k] ---
    if (TB) {
#pragma unroll
      for (int i = 0; i < 2; ++i) {
        const int c = tid + i * 256;
        const int n = c >> 2, k8 = (c & 3) << 3;
        async_copy_b128(lds_addr32(&Bs[n * LDS_K + k8]),
                        B + (size_t)(n0 + n) * ldb + k0 + k8);
      }
    } else {
      bf16x8 vb[2];
#pragma unroll
      for (int i = 0; i < 2; ++i) {
        const int c = tid + i * 256;
        const int k = c >> 4, n8 = (c & 15) << 3;
        vb[i] = *(const bf16x8*)(B + (size_t)(k0 + k) * ldb + n0 + n8);
      }
#pragma unroll
      for (int i = 0; i < 2; ++i) {
        const int c = tid + i * 256;
        const int k = c >> 4, n8 = (c & 15) << 3;
#pragma unroll
        for (int j = 0; j < 8; ++j) Bs[(n8 + j) * LDS_K + k] = vb[i][j];
      }
    }
    // prefetch next K tile into cache (global_prefetch_b8)
    if (k0 + BK < K) {
      const __bf16* pa = TA ? (A + (size_t)(k0 + BK + (tid >> 3)) * lda + m0 + (tid & 7) * 16)
                            : (A + (size_t)(m0 + (tid >> 1)) * lda + k0 + BK + (tid & 1) * 16);
      const __bf16* pb = TB ? (B + (size_t)(n0 + (tid >> 1)) * ldb + k0 + BK + (tid & 1) * 16)
                            : (B + (size_t)(k0 + BK + (tid >> 3)) * ldb + n0 + (tid & 7) * 16);
      __builtin_prefetch(pa, 0, 0);
      __builtin_prefetch(pb, 0, 0);
    }
    if (!TA || TB) wait_asynccnt0();   // async LDS writes visible before barrier
    __syncthreads();

    bf16x16 af[4], bfv[2];
#pragma unroll
    for (int mi = 0; mi < 4; ++mi)
      af[mi] = ld_afrag(&As[(wm * 64 + mi * 16 + lr) * LDS_K + kb]);
#pragma unroll
    for (int ni = 0; ni < 2; ++ni)
      bfv[ni] = ld_bfrag(&Bs[(wn * 32 + ni * 16 + lr) * LDS_K + kb]);

#pragma unroll
    for (int mi = 0; mi < 4; ++mi)
#pragma unroll
      for (int ni = 0; ni < 2; ++ni)
        acc[mi][ni] = __builtin_amdgcn_wmma_f32_16x16x32_bf16(
            false, af[mi], false, bfv[ni], (short)0, acc[mi][ni], false, false);

    __syncthreads();
  }

  // Epilogue: C/D f32 layout: VGPR i -> M = i (lanes<16) / 8+i (lanes>=16); N = lane&15.
  const int rhi = (lane >= 16) ? 8 : 0;
#pragma unroll
  for (int mi = 0; mi < 4; ++mi) {
#pragma unroll
    for (int ni = 0; ni < 2; ++ni) {
      const int col = n0 + wn * 32 + ni * 16 + lr;
#pragma unroll
      for (int i = 0; i < 8; ++i) {
        const int row = m0 + wm * 64 + mi * 16 + rhi + i;
        C[(size_t)row * ldc + col] = (OUT_T)acc[mi][ni][i];
      }
    }
  }
}

// ---------------------------------------------------------------------------
// Elementwise epilogue kernels (all over [B, N_P, H], grid.z = batch)
// ---------------------------------------------------------------------------
__global__ void build_cat4(const float* __restrict__ fw, const float* __restrict__ bw,
                           __bf16* __restrict__ cat) {
  const int b = blockIdx.z;
  const int idx = blockIdx.x * blockDim.x + threadIdx.x;   // < N_P*H
  const int h = idx & (H_DIM - 1);
  const int n = idx >> 8;
  const size_t o  = (size_t)b * N_P * H_DIM + idx;
  const size_t co = (size_t)b * N_P * 4 * H_DIM + (size_t)n * 4 * H_DIM + h;
  const float f = fw[o], w = bw[o];
  cat[co]             = (__bf16)f;
  cat[co + H_DIM]     = (__bf16)w;
  cat[co + 2 * H_DIM] = (__bf16)(f * w);
  cat[co + 3 * H_DIM] = (__bf16)(f - w);
}

__global__ void fuse_gate(const float* __restrict__ zf, const float* __restrict__ bias,
                          const float* __restrict__ fw, const float* __restrict__ bw,
                          const float* __restrict__ nf, float* __restrict__ agg,
                          __bf16* __restrict__ hx) {
  const int b = blockIdx.z;
  const int idx = blockIdx.x * blockDim.x + threadIdx.x;   // < N_P*H
  const int h = idx & (H_DIM - 1);
  const int n = idx >> 8;
  const size_t o = (size_t)b * N_P * H_DIM + idx;
  const float z = sigmoidf_(zf[o] + bias[h]);
  const float a = (1.0f - z) * fw[o] + z * bw[o];
  agg[o] = a;
  const size_t ho = (size_t)b * N_P * 2 * H_DIM + (size_t)n * 2 * H_DIM + h;
  hx[ho]         = (__bf16)nf[o];
  hx[ho + H_DIM] = (__bf16)a;
}

__global__ void build_rh(const float* __restrict__ rlin, const float* __restrict__ nf,
                         const float* __restrict__ agg, __bf16* __restrict__ rh) {
  const int b = blockIdx.z;
  const int idx = blockIdx.x * blockDim.x + threadIdx.x;   // < N_P*H
  const int h = idx & (H_DIM - 1);
  const int n = idx >> 8;
  const size_t o = (size_t)b * N_P * H_DIM + idx;
  const float r = sigmoidf_(rlin[o]);
  const size_t ho = (size_t)b * N_P * 2 * H_DIM + (size_t)n * 2 * H_DIM + h;
  rh[ho]         = (__bf16)(r * nf[o]);
  rh[ho + H_DIM] = (__bf16)agg[o];
}

__global__ void gru_combine(const float* __restrict__ zlin, const float* __restrict__ tlin,
                            float* __restrict__ nf) {
  const int b = blockIdx.z;
  const int idx = blockIdx.x * blockDim.x + threadIdx.x;   // < N_P*H
  const size_t o = (size_t)b * N_P * H_DIM + idx;
  const float z = sigmoidf_(zlin[o]);
  const float t = tanhf(tlin[o]);
  nf[o] = (1.0f - z) * nf[o] + z * t;
}

__global__ void unpack_out(const float* __restrict__ nf, float* __restrict__ out) {
  const size_t idx = (size_t)blockIdx.x * blockDim.x + threadIdx.x;
  const size_t total = (size_t)BATCH * N_REAL * H_DIM;
  if (idx >= total) return;
  const int b = (int)(idx / ((size_t)N_REAL * H_DIM));
  const int rem = (int)(idx % ((size_t)N_REAL * H_DIM));
  const int n = rem / H_DIM;
  const int h = rem % H_DIM;
  out[idx] = nf[(size_t)b * N_P * H_DIM + (size_t)n * H_DIM + h];
}

// ---------------------------------------------------------------------------
// Host-side orchestration
// ---------------------------------------------------------------------------
static inline size_t align256(size_t x) { return (x + 255) & ~(size_t)255; }

extern "C" void kernel_launch(void* const* d_in, const int* in_sizes, int n_in,
                              void* d_out, int out_size, void* d_ws, size_t ws_size,
                              hipStream_t stream) {
  (void)in_sizes; (void)n_in; (void)out_size;

  const float* nf_in  = (const float*)d_in[0];  // [8,1000,256]
  const float* n2e_in = (const float*)d_in[1];  // [8,4000,1000]
  const float* e2n_in = (const float*)d_in[2];  // [8,1000,4000]
  const float* Wz_in  = (const float*)d_in[3];  // [256,512]
  const float* Wr_in  = (const float*)d_in[4];  // [256,512]
  const float* Wt_in  = (const float*)d_in[5];  // [256,512]
  const float* Wf_in  = (const float*)d_in[6];  // [256,1024]
  const float* bias   = (const float*)d_in[7];  // [256]
  float* out = (float*)d_out;

  // ---- workspace carve-out ----
  char* p = (char*)d_ws;
  char* p_end = p + ws_size;
  auto alloc = [&](size_t bytes) -> void* { void* r = (void*)p; p += align256(bytes); return r; };

  __bf16* n2e_bf = (__bf16*)alloc((size_t)BATCH * E_P * N_P * 2);       // [B,E_P,N_P]
  __bf16* e2n_bf = (__bf16*)alloc((size_t)BATCH * N_P * E_P * 2);       // [B,N_P,E_P]
  __bf16* nf_bf  = (__bf16*)alloc((size_t)BATCH * N_P * H_DIM * 2);
  __bf16* t1_bf  = (__bf16*)alloc((size_t)BATCH * E_P * H_DIM * 2);
  __bf16* t2_bf  = (__bf16*)alloc((size_t)BATCH * E_P * H_DIM * 2);
  float*  bw_f   = (float*) alloc((size_t)BATCH * N_P * H_DIM * 4);
  float*  fw_f   = (float*) alloc((size_t)BATCH * N_P * H_DIM * 4);
  __bf16* cat_bf = (__bf16*)alloc((size_t)BATCH * N_P * 4 * H_DIM * 2);
  float*  zf_f   = (float*) alloc((size_t)BATCH * N_P * H_DIM * 4);
  float*  agg_f  = (float*) alloc((size_t)BATCH * N_P * H_DIM * 4);
  __bf16* hx_bf  = (__bf16*)alloc((size_t)BATCH * N_P * 2 * H_DIM * 2);
  float*  zlin   = (float*) alloc((size_t)BATCH * N_P * H_DIM * 4);
  float*  rlin   = (float*) alloc((size_t)BATCH * N_P * H_DIM * 4);
  __bf16* rh_bf  = (__bf16*)alloc((size_t)BATCH * N_P * 2 * H_DIM * 2);
  float*  tlin   = (float*) alloc((size_t)BATCH * N_P * H_DIM * 4);
  float*  nf_cur = (float*) alloc((size_t)BATCH * N_P * H_DIM * 4);
  __bf16* Wz_bf  = (__bf16*)alloc((size_t)H_DIM * 2 * H_DIM * 2);
  __bf16* Wr_bf  = (__bf16*)alloc((size_t)H_DIM * 2 * H_DIM * 2);
  __bf16* Wt_bf  = (__bf16*)alloc((size_t)H_DIM * 2 * H_DIM * 2);
  __bf16* Wf_bf  = (__bf16*)alloc((size_t)H_DIM * 4 * H_DIM * 2);
  if (p > p_end) return;  // insufficient workspace

  const dim3 blk(256);
  const dim3 grid_nh(N_P * H_DIM / 256, 1, BATCH);

  // ---- one-time bf16 conversion (padded) ----
  pack_bf16<<<dim3((size_t)E_P * N_P / 256, 1, BATCH), blk, 0, stream>>>(
      n2e_in, n2e_bf, E_REAL, N_REAL, E_P, N_P);
  pack_bf16<<<dim3((size_t)N_P * E_P / 256, 1, BATCH), blk, 0, stream>>>(
      e2n_in, e2n_bf, N_REAL, E_REAL, N_P, E_P);
  pack_f32<<<grid_nh, blk, 0, stream>>>(nf_in, nf_cur, N_REAL, H_DIM, N_P, H_DIM);
  pack_bf16<<<dim3(H_DIM * 2 * H_DIM / 256, 1, 1), blk, 0, stream>>>(Wz_in, Wz_bf, H_DIM, 2 * H_DIM, H_DIM, 2 * H_DIM);
  pack_bf16<<<dim3(H_DIM * 2 * H_DIM / 256, 1, 1), blk, 0, stream>>>(Wr_in, Wr_bf, H_DIM, 2 * H_DIM, H_DIM, 2 * H_DIM);
  pack_bf16<<<dim3(H_DIM * 2 * H_DIM / 256, 1, 1), blk, 0, stream>>>(Wt_in, Wt_bf, H_DIM, 2 * H_DIM, H_DIM, 2 * H_DIM);
  pack_bf16<<<dim3(H_DIM * 4 * H_DIM / 256, 1, 1), blk, 0, stream>>>(Wf_in, Wf_bf, H_DIM, 4 * H_DIM, H_DIM, 4 * H_DIM);

  const dim3 grid_EH(H_DIM / BN, E_P / BM, BATCH);   // 2 x 32 x 8
  const dim3 grid_NH(H_DIM / BN, N_P / BM, BATCH);   // 2 x  8 x 8

  for (int hop = 0; hop < 2; ++hop) {
    // nf -> bf16
    pack_bf16<<<grid_nh, blk, 0, stream>>>(nf_cur, nf_bf, N_P, H_DIM, N_P, H_DIM);

    // t1 = node2edge @ nf                  [E_P,N_P] x [N_P,H]
    gemm_wmma<false, false, __bf16><<<grid_EH, blk, 0, stream>>>(
        n2e_bf, nf_bf, t1_bf, E_P, H_DIM, N_P, N_P, H_DIM, H_DIM,
        (size_t)E_P * N_P, (size_t)N_P * H_DIM, (size_t)E_P * H_DIM);
    // t2 = edge2node^T @ nf                A^T with A=[N_P,E_P]
    gemm_wmma<true, false, __bf16><<<grid_EH, blk, 0, stream>>>(
        e2n_bf, nf_bf, t2_bf, E_P, H_DIM, N_P, E_P, H_DIM, H_DIM,
        (size_t)N_P * E_P, (size_t)N_P * H_DIM, (size_t)E_P * H_DIM);
    // bw = edge2node @ t1                  [N_P,E_P] x [E_P,H]
    gemm_wmma<false, false, float><<<grid_NH, blk, 0, stream>>>(
        e2n_bf, t1_bf, bw_f, N_P, H_DIM, E_P, E_P, H_DIM, H_DIM,
        (size_t)N_P * E_P, (size_t)E_P * H_DIM, (size_t)N_P * H_DIM);
    // fw = node2edge^T @ t2                A^T with A=[E_P,N_P]
    gemm_wmma<true, false, float><<<grid_NH, blk, 0, stream>>>(
        n2e_bf, t2_bf, fw_f, N_P, H_DIM, E_P, N_P, H_DIM, H_DIM,
        (size_t)E_P * N_P, (size_t)E_P * H_DIM, (size_t)N_P * H_DIM);

    // gated fusion: cat = [fw, bw, fw*bw, fw-bw]; z = sigmoid(cat @ Wf^T + bf)
    build_cat4<<<grid_nh, blk, 0, stream>>>(fw_f, bw_f, cat_bf);
    gemm_wmma<false, true, float><<<grid_NH, blk, 0, stream>>>(
        cat_bf, Wf_bf, zf_f, N_P, H_DIM, 4 * H_DIM, 4 * H_DIM, 4 * H_DIM, H_DIM,
        (size_t)N_P * 4 * H_DIM, 0, (size_t)N_P * H_DIM);
    fuse_gate<<<grid_nh, blk, 0, stream>>>(zf_f, bias, fw_f, bw_f, nf_cur, agg_f, hx_bf);

    // GRU: z/r gates, candidate, combine
    gemm_wmma<false, true, float><<<grid_NH, blk, 0, stream>>>(
        hx_bf, Wz_bf, zlin, N_P, H_DIM, 2 * H_DIM, 2 * H_DIM, 2 * H_DIM, H_DIM,
        (size_t)N_P * 2 * H_DIM, 0, (size_t)N_P * H_DIM);
    gemm_wmma<false, true, float><<<grid_NH, blk, 0, stream>>>(
        hx_bf, Wr_bf, rlin, N_P, H_DIM, 2 * H_DIM, 2 * H_DIM, 2 * H_DIM, H_DIM,
        (size_t)N_P * 2 * H_DIM, 0, (size_t)N_P * H_DIM);
    build_rh<<<grid_nh, blk, 0, stream>>>(rlin, nf_cur, agg_f, rh_bf);
    gemm_wmma<false, true, float><<<grid_NH, blk, 0, stream>>>(
        rh_bf, Wt_bf, tlin, N_P, H_DIM, 2 * H_DIM, 2 * H_DIM, 2 * H_DIM, H_DIM,
        (size_t)N_P * 2 * H_DIM, 0, (size_t)N_P * H_DIM);
    gru_combine<<<grid_nh, blk, 0, stream>>>(zlin, tlin, nf_cur);
  }

  // strip padding -> d_out [8,1000,256]
  const size_t out_total = (size_t)BATCH * N_REAL * H_DIM;
  unpack_out<<<dim3((out_total + 255) / 256), blk, 0, stream>>>(nf_cur, out);
}
